// GAT_17308718203315
// MI455X (gfx1250) — matrix operand order repaired and checked by reading, not compiled, not run on hardware
//
#include <hip/hip_runtime.h>
#include <hip/hip_bf16.h>

// Problem constants (from reference)
#define DD 256   // dim*heads
#define HH 4
#define CC 64
#define LL 3
#define GG 25

typedef __attribute__((ext_vector_type(16))) __bf16       bf16x16;
typedef __attribute__((ext_vector_type(8)))  float        f32x8;
typedef __attribute__((ext_vector_type(4)))  unsigned int u32x4;

union Frag16 { bf16x16 v; unsigned short s[16]; u32x4 q[2]; };

// float -> bf16 (round to nearest even), bit-exact and no __bf16 cast needed
__device__ __forceinline__ unsigned short f2b(float f) {
  unsigned int u = __float_as_uint(f);
  unsigned int r = u + 0x7fffu + ((u >> 16) & 1u);
  return (unsigned short)(r >> 16);
}

// order-preserving float<->uint encoding for atomicMax-based segment max
__device__ __forceinline__ unsigned int encodeF(float f) {
  unsigned int u = __float_as_uint(f);
  return (u & 0x80000000u) ? ~u : (u | 0x80000000u);
}
__device__ __forceinline__ float decodeF(unsigned int e) {
  return (e & 0x80000000u) ? __uint_as_float(e & 0x7fffffffu)
                           : __uint_as_float(~e);
}

// CDNA5 async global->LDS DMA (ASYNCcnt tracked); lds_off = wave-relative byte
// offset (low 32 bits of a generic pointer to __shared__ are the LDS offset).
__device__ __forceinline__ void async_ld16(unsigned lds_off, const void* gptr) {
  asm volatile("global_load_async_to_lds_b128 %0, %1, off"
               :: "v"(lds_off), "v"(gptr) : "memory");
}
__device__ __forceinline__ void wait_async0() {
  asm volatile("s_wait_asynccnt 0x0" ::: "memory");
}

// ---------------------------------------------------------------------------
// WMMA GEMM: C[M,256] = A[M,256](f32->bf16) * W[256,256] (+bias)
// W supplied pre-transposed+bf16: WT[n*256+k]. 8 waves/block, 16 rows/wave.
// Double-buffered 32KB LDS chunks filled by async DMA; XOR bank swizzle.
// ---------------------------------------------------------------------------
__device__ __forceinline__ void stage_chunk_async(const unsigned short* __restrict__ src,
                                                  unsigned ldsBase, int tid)
{
  // 64 rows x 256 bf16 = 2048 x 16B; granule = 32B, swizzled by (row & 15)
#pragma unroll
  for (int it = 0; it < 8; ++it) {
    int i = tid + it * 256;
    int n = i >> 5;                 // row (output col) 0..63
    int j = i & 31;                 // 16B unit within row
    int g = j >> 1;                 // 32B granule 0..15
    int p = j & 1;                  // half of granule
    int gs = g ^ (n & 15);          // bank swizzle
    unsigned dst = ldsBase + (unsigned)(n * 512 + gs * 32 + p * 16);
    async_ld16(dst, (const void*)((const char*)src + (size_t)i * 16));
  }
}

__global__ __launch_bounds__(256)
void gemm_nodes(const float* __restrict__ A, const unsigned short* __restrict__ WT,
                const float* __restrict__ bias, float* __restrict__ C, int M)
{
  __shared__ unsigned short lw[2][64 * 256];        // 2 x 32KB double buffer
  const int tid  = threadIdx.x;
  const int wave = tid >> 5;
  const int lane = tid & 31;
  const int rowBase = blockIdx.x * 128 + wave * 16;
  const int khalf   = (lane >> 4) * 16;             // A-frag K-half select
  const int nl      = lane & 15;

  int m = rowBase + nl;
  int msafe = (m < M) ? m : (M - 1);
  const float* arow = A + (size_t)msafe * DD;

  // kick off DMA of chunk 0 while we load/convert A
  const unsigned ldsOff0 = (unsigned)(uintptr_t)(&lw[0][0]);
  const unsigned ldsOff1 = (unsigned)(uintptr_t)(&lw[1][0]);
  stage_chunk_async(WT, ldsOff0, tid);

  // Load + convert all 8 A fragments (K = 256 = 8 * 32), keep in registers
  Frag16 a[8];
#pragma unroll
  for (int kk = 0; kk < 8; ++kk) {
    const float* p0 = arow + kk * 32 + khalf;
#pragma unroll
    for (int j = 0; j < 8; ++j) a[kk].s[j]     = f2b(p0[j]);        // K = khalf+j
#pragma unroll
    for (int j = 0; j < 8; ++j) a[kk].s[8 + j] = f2b(p0[8 + j]);    // K = khalf+8+j
  }

  wait_async0();
  __syncthreads();                                   // chunk 0 resident

  for (int nc = 0; nc < 4; ++nc) {                   // 4 chunks of 64 output cols
    const unsigned short* cur = &lw[nc & 1][0];
    if (nc < 3)                                      // overlap DMA of next chunk
      stage_chunk_async(WT + (size_t)(nc + 1) * 64 * 256,
                        (nc & 1) ? ldsOff0 : ldsOff1, tid);

    f32x8 acc[4];
#pragma unroll
    for (int nt = 0; nt < 4; ++nt)
#pragma unroll
      for (int v = 0; v < 8; ++v) acc[nt][v] = 0.0f;

#pragma unroll
    for (int kk = 0; kk < 8; ++kk) {
      Frag16 b[4];
#pragma unroll
      for (int nt = 0; nt < 4; ++nt) {
        // B frag: N = nt*16 + nl, K = kk*32 + (lane>=16 ? 16:0) + j
        int nloc = nt * 16 + nl;
        int g    = kk * 2 + (lane >> 4);             // 32B granule index
        int gs   = g ^ (nloc & 15);                  // undo bank swizzle
        const u32x4* bp = (const u32x4*)((const char*)cur + nloc * 512 + gs * 32);
        b[nt].q[0] = bp[0];
        b[nt].q[1] = bp[1];
      }
#pragma unroll
      for (int nt = 0; nt < 4; ++nt)
        acc[nt] = __builtin_amdgcn_wmma_f32_16x16x32_bf16(
            false, a[kk].v, false, b[nt].v, (short)0, acc[nt], false, false);
    }

    // epilogue: C layout — lane n = nl, rows m = rowBase + v + 8*(lane>=16)
    const int mb = rowBase + ((lane >> 4) * 8);
#pragma unroll
    for (int nt = 0; nt < 4; ++nt) {
      int n = nc * 64 + nt * 16 + nl;
      float bv = bias ? bias[n] : 0.0f;
#pragma unroll
      for (int v = 0; v < 8; ++v) {
        int mm = mb + v;
        if (mm < M) C[(size_t)mm * DD + n] = acc[nt][v] + bv;
      }
    }
    wait_async0();                                   // own DMA of next chunk done
    __syncthreads();                                 // everyone's DMA done + buffer free
  }
}

// ---------------------------------------------------------------------------
// Small prep: Mdst[l][k][h] = sum_c W_dst[l][k][h*64+c]*att_dst[l][h][c]
//             Aed [l][d][h] = sum_c W_edge[l][d][h*64+c]*att_edge[l][h][c]
// ---------------------------------------------------------------------------
__global__ void prep_small(const float* __restrict__ W_dst, const float* __restrict__ att_dst,
                           const float* __restrict__ W_edge, const float* __restrict__ att_edge,
                           float* __restrict__ Mdst, float* __restrict__ Aed)
{
  int id = blockIdx.x * 256 + threadIdx.x;
  if (id < LL * 256 * 4) {
    int l = id >> 10, rem = id & 1023, k = rem >> 2, hh = rem & 3;
    const float* w = W_dst + ((size_t)l * 256 + k) * 256 + hh * 64;
    const float* a = att_dst + l * 256 + hh * 64;
    float sv = 0.f;
    for (int c = 0; c < 64; ++c) sv += w[c] * a[c];
    Mdst[id] = sv;                                  // [l][k][h]
  } else if (id < LL * 256 * 4 + LL * 2 * 4) {
    int j = id - LL * 1024;
    int l = j >> 3, d = (j >> 2) & 1, hh = j & 3;
    const float* w = W_edge + ((size_t)l * 2 + d) * 256 + hh * 64;
    const float* a = att_edge + l * 256 + hh * 64;
    float sv = 0.f;
    for (int c = 0; c < 64; ++c) sv += w[c] * a[c];
    Aed[j] = sv;                                    // [l][d][h]
  }
}

// transpose + convert W_pre and W_src[0..2] to bf16 WT[mat][n*256+k]
__global__ void conv_wt(const float* __restrict__ W_pre, const float* __restrict__ W_src,
                        unsigned short* __restrict__ WT)
{
  int id = blockIdx.x * 256 + threadIdx.x;          // 4*65536 total
  if (id >= 4 * 65536) return;
  int mat = id >> 16, rem = id & 65535;
  int n = rem >> 8, k = rem & 255;
  const float* W = (mat == 0) ? W_pre : (W_src + (size_t)(mat - 1) * 65536);
  WT[id] = f2b(W[k * 256 + n]);
}

// ---------------------------------------------------------------------------
// Per-node attention dots: s_src[n,h]=sum_c xs*att_src ; s_dst[n,h]=h@Mdst
// One wave per node; lane owns 8 consecutive k.
// ---------------------------------------------------------------------------
__global__ __launch_bounds__(256)
void sdot_kernel(const float* __restrict__ xs, const float* __restrict__ h,
                 const float* __restrict__ att_src_l, const float* __restrict__ Mdst_l,
                 float* __restrict__ s_src, float* __restrict__ s_dst, int Nn)
{
  int wave = threadIdx.x >> 5, lane = threadIdx.x & 31;
  int n = blockIdx.x * 8 + wave;
  if (n >= Nn) return;
  const float* xr = xs + (size_t)n * DD + lane * 8;
  const float* hr = h  + (size_t)n * DD + lane * 8;
  const float* av = att_src_l + lane * 8;
  const float* md = Mdst_l + lane * 8 * 4;
  float ss = 0.f, sd0 = 0.f, sd1 = 0.f, sd2 = 0.f, sd3 = 0.f;
#pragma unroll
  for (int j = 0; j < 8; ++j) {
    ss += xr[j] * av[j];
    float hv = hr[j];
    sd0 += hv * md[j * 4 + 0]; sd1 += hv * md[j * 4 + 1];
    sd2 += hv * md[j * 4 + 2]; sd3 += hv * md[j * 4 + 3];
  }
  for (int o = 1; o < 8; o <<= 1) ss += __shfl_xor(ss, o, 32);      // per-head group of 8
  for (int o = 1; o < 32; o <<= 1) {
    sd0 += __shfl_xor(sd0, o, 32); sd1 += __shfl_xor(sd1, o, 32);
    sd2 += __shfl_xor(sd2, o, 32); sd3 += __shfl_xor(sd3, o, 32);
  }
  if ((lane & 7) == 0) s_src[(size_t)n * 4 + (lane >> 3)] = ss;
  if (lane == 0) {
    s_dst[(size_t)n * 4 + 0] = sd0; s_dst[(size_t)n * 4 + 1] = sd1;
    s_dst[(size_t)n * 4 + 2] = sd2; s_dst[(size_t)n * 4 + 3] = sd3;
  }
}

// edge pass 1: logits + leaky-relu + segment max (encoded atomicMax)
__global__ void edge_logits(const int* __restrict__ ei, const float* __restrict__ eattr,
                            const float* __restrict__ s_src, const float* __restrict__ s_dst,
                            const float* __restrict__ Aed_l, float* __restrict__ abuf,
                            unsigned int* __restrict__ mbuf, int E)
{
  int e = blockIdx.x * 256 + threadIdx.x;
  if (e >= E) return;
  int r = ei[e], c = ei[E + e];
  float e0 = eattr[(size_t)e * 2], e1 = eattr[(size_t)e * 2 + 1];
#pragma unroll
  for (int hh = 0; hh < 4; ++hh) {
    float a = s_src[(size_t)r * 4 + hh] + s_dst[(size_t)c * 4 + hh]
            + e0 * Aed_l[hh] + e1 * Aed_l[4 + hh];
    a = (a > 0.f) ? a : 0.2f * a;
    abuf[(size_t)e * 4 + hh] = a;
    atomicMax(&mbuf[(size_t)c * 4 + hh], encodeF(a));
  }
}

// edge pass 2: w = exp(a - m[col]); z[col] += w
__global__ void edge_exp(const int* __restrict__ ei, float* __restrict__ abuf,
                         const unsigned int* __restrict__ mbuf, float* __restrict__ zbuf, int E)
{
  int e = blockIdx.x * 256 + threadIdx.x;
  if (e >= E) return;
  int c = ei[E + e];
#pragma unroll
  for (int hh = 0; hh < 4; ++hh) {
    float m = decodeF(mbuf[(size_t)c * 4 + hh]);
    float w = __expf(abuf[(size_t)e * 4 + hh] - m);
    abuf[(size_t)e * 4 + hh] = w;
    atomicAdd(&zbuf[(size_t)c * 4 + hh], w);
  }
}

// edge pass 3: out[col] += xs[row] * alpha ; 64 threads per edge, float4 each
__global__ __launch_bounds__(256)
void edge_scatter(const int* __restrict__ ei, const float* __restrict__ abuf,
                  const float* __restrict__ zbuf, const float* __restrict__ xs,
                  float* __restrict__ out, int E)
{
  long long gid = (long long)blockIdx.x * 256 + threadIdx.x;
  int e = (int)(gid >> 6);
  int t = (int)(gid & 63);
  if (e >= E) return;
  int r = ei[e], c = ei[E + e];
  int hh = t >> 4;                                   // 16 threads (64 ch) per head
  float alpha = abuf[(size_t)e * 4 + hh] / (zbuf[(size_t)c * 4 + hh] + 1e-16f);
  const float* xp = xs + (size_t)r * DD + t * 4;
  float* op = out + (size_t)c * DD + t * 4;
  atomicAdd(op + 0, xp[0] * alpha);
  atomicAdd(op + 1, xp[1] * alpha);
  atomicAdd(op + 2, xp[2] * alpha);
  atomicAdd(op + 3, xp[3] * alpha);
}

// LayerNorm + PReLU (in place in out) + residual into h. One wave per node.
__global__ __launch_bounds__(256)
void ln_prelu(float* __restrict__ out, float* __restrict__ h,
              const float* __restrict__ gam, const float* __restrict__ bet,
              const float* __restrict__ pa, int Nn)
{
  int wave = threadIdx.x >> 5, lane = threadIdx.x & 31;
  int n = blockIdx.x * 8 + wave;
  if (n >= Nn) return;
  float* orow = out + (size_t)n * DD;
  float* hrow = h   + (size_t)n * DD;
  float v[8], s = 0.f, sq = 0.f;
#pragma unroll
  for (int j = 0; j < 8; ++j) {
    float x = orow[lane + j * 32];
    v[j] = x; s += x; sq += x * x;
  }
  for (int o = 1; o < 32; o <<= 1) { s += __shfl_xor(s, o, 32); sq += __shfl_xor(sq, o, 32); }
  float mu = s * (1.f / 256.f);
  float var = sq * (1.f / 256.f) - mu * mu;
  float inv = rsqrtf(var + 1e-5f);
#pragma unroll
  for (int j = 0; j < 8; ++j) {
    int d = lane + j * 32;
    float x = (v[j] - mu) * inv * gam[d] + bet[d];
    x = (x > 0.f) ? x : pa[d] * x;
    orow[d] = x;
    hrow[d] += x;
  }
}

// segment-sum over sorted batch: 64 nodes/block, thread = channel, run-length flush
__global__ __launch_bounds__(256)
void seg_accum(const float* __restrict__ v, const int* __restrict__ batch,
               float* __restrict__ sums, int Nn)
{
  int d = threadIdx.x;
  int n0 = blockIdx.x * 64;
  int n1 = n0 + 64; if (n1 > Nn) n1 = Nn;
  int gcur = -1; float acc = 0.f;
  for (int n = n0; n < n1; ++n) {
    int g = batch[n];
    if (g != gcur) {
      if (gcur >= 0) atomicAdd(&sums[(size_t)gcur * DD + d], acc);
      gcur = g; acc = 0.f;
    }
    acc += v[(size_t)n * DD + d];
  }
  if (gcur >= 0) atomicAdd(&sums[(size_t)gcur * DD + d], acc);
}

__global__ void count_kernel(const int* __restrict__ batch, float* __restrict__ cnt, int Nn) {
  int n = blockIdx.x * 256 + threadIdx.x;
  if (n < Nn) atomicAdd(&cnt[batch[n]], 1.0f);
}

__global__ void zero_f(float* __restrict__ p, int n) {
  int i = blockIdx.x * 256 + threadIdx.x;
  if (i < n) p[i] = 0.f;
}

__global__ void init_out(float* __restrict__ out, const float* __restrict__ bias_l, long long total) {
  long long i = (long long)blockIdx.x * 256 + threadIdx.x;
  if (i < total) out[i] = bias_l[i & 255];
}

__global__ void init_att(unsigned int* __restrict__ mbuf, float* __restrict__ zbuf, int Nn) {
  int i = blockIdx.x * 256 + threadIdx.x;
  if (i < Nn * 4) { mbuf[i] = 0u; zbuf[i] = 0.f; }   // 0 encodes below any real value
}

// final head: per-graph means -> MLP -> risk. One block per graph.
__global__ __launch_bounds__(256)
void post_mlp(const float* __restrict__ sums, const float* __restrict__ cnt,
              const float* __restrict__ W1, const float* __restrict__ b1,
              const float* __restrict__ W2, const float* __restrict__ b2,
              const float* __restrict__ Wr, const float* __restrict__ br,
              float* __restrict__ outp)
{
  __shared__ float xc[1024];
  __shared__ float p1[256];
  __shared__ float red[256];
  int g = blockIdx.x, t = threadIdx.x;
  float c = fmaxf(cnt[g], 1.0f);
#pragma unroll
  for (int s = 0; s < LL + 1; ++s)
    xc[s * 256 + t] = sums[((size_t)s * GG + g) * DD + t] / c;
  __syncthreads();
  float acc = b1[t];
  for (int k = 0; k < 1024; ++k) acc += xc[k] * W1[(size_t)k * 256 + t];
  p1[t] = fmaxf(acc, 0.f);
  __syncthreads();
  float acc2 = b2[t];
  for (int k = 0; k < 256; ++k) acc2 += p1[k] * W2[(size_t)k * 256 + t];
  red[t] = fmaxf(acc2, 0.f) * Wr[t];
  __syncthreads();
  for (int o = 128; o > 0; o >>= 1) { if (t < o) red[t] += red[t + o]; __syncthreads(); }
  if (t == 0) outp[g] = red[0] + br[0];
}

// ---------------------------------------------------------------------------
extern "C" void kernel_launch(void* const* d_in, const int* in_sizes, int n_in,
                              void* d_out, int out_size, void* d_ws, size_t ws_size,
                              hipStream_t stream)
{
  (void)n_in; (void)out_size; (void)ws_size;
  const float* x         = (const float*)d_in[0];
  const float* edge_attr = (const float*)d_in[1];
  const int*   ei        = (const int*)d_in[2];
  const int*   batch     = (const int*)d_in[3];
  const float* W_pre     = (const float*)d_in[4];
  const float* b_pre     = (const float*)d_in[5];
  const float* W_src     = (const float*)d_in[6];
  const float* W_dst     = (const float*)d_in[7];
  const float* W_edge    = (const float*)d_in[8];
  const float* att_src   = (const float*)d_in[9];
  const float* att_dst   = (const float*)d_in[10];
  const float* att_edge  = (const float*)d_in[11];
  const float* bias_conv = (const float*)d_in[12];
  const float* ln_gamma  = (const float*)d_in[13];
  const float* ln_beta   = (const float*)d_in[14];
  const float* prelu_a   = (const float*)d_in[15];
  const float* W_post1   = (const float*)d_in[16];
  const float* b_post1   = (const float*)d_in[17];
  const float* W_post2   = (const float*)d_in[18];
  const float* b_post2   = (const float*)d_in[19];
  const float* W_risk    = (const float*)d_in[20];
  const float* b_risk    = (const float*)d_in[21];

  const int Nn = in_sizes[0] / DD;   // x: [N, 256]
  const int E  = in_sizes[1] / 2;    // edge_attr: [E, 2]

  char* ws = (char*)d_ws;
  size_t off = 0;
  auto wsAlloc = [&](size_t nbytes) -> void* {
    void* p = (void*)(ws + off);
    off = (off + nbytes + 255) & ~(size_t)255;
    return p;
  };
  float*          hbuf  = (float*)wsAlloc((size_t)Nn * DD * 4);
  float*          xsbuf = (float*)wsAlloc((size_t)Nn * DD * 4);
  float*          obuf  = (float*)wsAlloc((size_t)Nn * DD * 4);
  unsigned short* wtb   = (unsigned short*)wsAlloc((size_t)4 * 65536 * 2);
  float*          ssrc  = (float*)wsAlloc((size_t)Nn * 4 * 4);
  float*          sdst  = (float*)wsAlloc((size_t)Nn * 4 * 4);
  float*          abuf  = (float*)wsAlloc((size_t)E * 4 * 4);
  unsigned int*   mbuf  = (unsigned int*)wsAlloc((size_t)Nn * 4 * 4);
  float*          zbuf  = (float*)wsAlloc((size_t)Nn * 4 * 4);
  float*          Mdst  = (float*)wsAlloc((size_t)LL * 256 * 4 * 4);
  float*          Aed   = (float*)wsAlloc((size_t)LL * 2 * 4 * 4);
  float*          gsums = (float*)wsAlloc((size_t)(LL + 1) * GG * DD * 4);
  float*          cnt   = (float*)wsAlloc((size_t)GG * 4);

  zero_f<<<((LL + 1) * GG * DD + 255) / 256, 256, 0, stream>>>(gsums, (LL + 1) * GG * DD);
  zero_f<<<1, 256, 0, stream>>>(cnt, GG);
  count_kernel<<<(Nn + 255) / 256, 256, 0, stream>>>(batch, cnt, Nn);
  conv_wt<<<1024, 256, 0, stream>>>(W_pre, W_src, wtb);
  prep_small<<<13, 256, 0, stream>>>(W_dst, att_dst, W_edge, att_edge, Mdst, Aed);

  const int gemmBlocks = (Nn + 127) / 128;
  // h = x @ W_pre + b_pre
  gemm_nodes<<<gemmBlocks, 256, 0, stream>>>(x, wtb, b_pre, hbuf, Nn);
  seg_accum<<<(Nn + 63) / 64, 256, 0, stream>>>(hbuf, batch, gsums, Nn);

  for (int l = 0; l < LL; ++l) {
    gemm_nodes<<<gemmBlocks, 256, 0, stream>>>(hbuf, wtb + (size_t)(1 + l) * 65536,
                                               nullptr, xsbuf, Nn);
    sdot_kernel<<<(Nn + 7) / 8, 256, 0, stream>>>(xsbuf, hbuf, att_src + l * DD,
                                                  Mdst + l * 1024, ssrc, sdst, Nn);
    init_att<<<(Nn * 4 + 255) / 256, 256, 0, stream>>>(mbuf, zbuf, Nn);
    init_out<<<(int)(((long long)Nn * DD + 255) / 256), 256, 0, stream>>>(
        obuf, bias_conv + l * DD, (long long)Nn * DD);
    edge_logits<<<(E + 255) / 256, 256, 0, stream>>>(ei, edge_attr, ssrc, sdst,
                                                     Aed + l * 8, abuf, mbuf, E);
    edge_exp<<<(E + 255) / 256, 256, 0, stream>>>(ei, abuf, mbuf, zbuf, E);
    edge_scatter<<<(int)(((long long)E * 64 + 255) / 256), 256, 0, stream>>>(
        ei, abuf, zbuf, xsbuf, obuf, E);
    ln_prelu<<<(Nn + 7) / 8, 256, 0, stream>>>(obuf, hbuf, ln_gamma + l * DD,
                                               ln_beta + l * DD, prelu_a + l * DD, Nn);
    seg_accum<<<(Nn + 63) / 64, 256, 0, stream>>>(obuf, batch,
                                                  gsums + (size_t)(l + 1) * GG * DD, Nn);
  }

  post_mlp<<<GG, 256, 0, stream>>>(gsums, cnt, W_post1, b_post1, W_post2, b_post2,
                                   W_risk, b_risk, (float*)d_out);
}
